// RNNModel_73212012528166
// MI455X (gfx1250) — matrix operand order repaired and checked by reading, not compiled, not run on hardware
//
#include <hip/hip_runtime.h>
#include <hip/hip_bf16.h>
#include <stdint.h>

#define NBASIS  3
#define NHID    1024
#define NTOKEN  50000
#define NSEMEME 2000
#define NSENSE  50000
#define NB      256      // SEQ*BSZ
#define KSEM    4

typedef __attribute__((ext_vector_type(16))) __bf16 v16bf;
typedef __attribute__((ext_vector_type(8)))  float  v8f;
typedef unsigned short u16;

union AFrag { uint4 q[2]; v16bf v; };
union CFrag { v8f v; float f[8]; };

// round f32 to bf16 (RNE) in the integer domain; result lives in bits [31:16]
static __device__ __forceinline__ unsigned rnd_bf(float x) {
  unsigned u = __float_as_uint(x);
  return u + 0x7FFFu + ((u >> 16) & 1u);
}

static __device__ __forceinline__ u16 f2bf(float x) {
  return (u16)(rnd_bf(x) >> 16);
}

// pack two floats into one dword of bf16 (lo -> [15:0], hi -> [31:16])
// one v_perm_b32 merges the two rounded high halves
static __device__ __forceinline__ unsigned pk2(float lo, float hi) {
  return __builtin_amdgcn_perm(rnd_bf(hi), rnd_bf(lo), 0x07060302u);
}

// convert 16 consecutive f32 -> 16 bf16, store as two uint4 (32B, aligned)
static __device__ __forceinline__ void cvt16_store(const float4* __restrict__ s4,
                                                   u16* __restrict__ dst) {
  float4 v0 = s4[0], v1 = s4[1], v2 = s4[2], v3 = s4[3];
  uint4 lo = {pk2(v0.x, v0.y), pk2(v0.z, v0.w), pk2(v1.x, v1.y), pk2(v1.z, v1.w)};
  uint4 hi = {pk2(v2.x, v2.y), pk2(v2.z, v2.w), pk2(v3.x, v3.y), pk2(v3.z, v3.w)};
  ((uint4*)dst)[0] = lo;
  ((uint4*)dst)[1] = hi;
}

static __device__ __forceinline__ v8f wmma_bf16(const v16bf& a, const v16bf& b, const v8f& c) {
  return __builtin_amdgcn_wmma_f32_16x16x32_bf16(false, a, false, b, (short)0, c, false, false);
}

// CDNA5 async global->LDS copy of 32 bytes (two b128), per-lane addresses
static __device__ __forceinline__ void async_copy32(unsigned lds_off, const u16* gsrc) {
  asm volatile(
      "global_load_async_to_lds_b128 %0, %1, off\n\t"
      "global_load_async_to_lds_b128 %0, %1, off offset:16"
      :: "v"(lds_off), "v"(gsrc) : "memory");
}
static __device__ __forceinline__ void wait_async0() {
  asm volatile("s_wait_asynccnt 0x0" ::: "memory");
}

// ---------------------------------------------------------------- x -> bf16 (once)
__global__ __launch_bounds__(256)
void xbf_kernel(const float* __restrict__ x, u16* __restrict__ xbf) {
  int i = (blockIdx.x * 256 + threadIdx.x) * 16;
  cvt16_store((const float4*)(x + i), xbf + i);
}

// ---------------------------------------------------------------- wsm = softmax_k(multi_weight)
__global__ __launch_bounds__(256)
void wsm_kernel(const float* __restrict__ mw, float* __restrict__ wsm) {
  int m = blockIdx.x * 256 + threadIdx.x;
  if (m >= NSEMEME) return;
  float a0 = mw[0 * NSEMEME + m], a1 = mw[1 * NSEMEME + m], a2 = mw[2 * NSEMEME + m];
  float mx = fmaxf(a0, fmaxf(a1, a2));
  float e0 = expf(a0 - mx), e1 = expf(a1 - mx), e2 = expf(a2 - mx);
  float inv = 1.0f / (e0 + e1 + e2);
  wsm[m * 3 + 0] = e0 * inv;
  wsm[m * 3 + 1] = e1 * inv;
  wsm[m * 3 + 2] = e2 * inv;
}

// ---------------------------------------------------------------- gate = sigmoid(x @ Wg^T + b)
__global__ __launch_bounds__(256)
void gate_kernel(const u16* __restrict__ xbf, const float* __restrict__ wg,
                 const float* __restrict__ gbias, float* __restrict__ gate) {
  __shared__ u16 As[128 * 32];
  __shared__ u16 Bs[128 * 32];
  const int tid = threadIdx.x, lane = tid & 31, wid = tid >> 5;
  const int waveM = (wid & 3) * 32, waveN = (wid >> 2) * 64;
  const int blockM = blockIdx.y * 128, blockN = blockIdx.x * 128;
  const int r = tid >> 1, c0 = (tid & 1) * 16;
  const unsigned ldsa = (unsigned)(uintptr_t)(As + r * 32 + c0);
  const u16* arow = xbf + (size_t)(blockM + r) * NHID + c0;

  CFrag acc[2][4];
#pragma unroll
  for (int tm = 0; tm < 2; ++tm)
#pragma unroll
    for (int tn = 0; tn < 4; ++tn)
#pragma unroll
      for (int i = 0; i < 8; ++i) acc[tm][tn].f[i] = 0.0f;

  for (int k0 = 0; k0 < NHID; k0 += 32) {
    // stage A: async bf16 copy
    async_copy32(ldsa, arow + k0);
    // stage B (Wg rows, f32 -> bf16)
    {
      int m = blockN + r;
      u16* dst = Bs + r * 32 + c0;
      if (m < NSEMEME) {
        cvt16_store((const float4*)(wg + (size_t)m * NHID + k0 + c0), dst);
      } else {
        uint4 z = {0, 0, 0, 0};
        ((uint4*)dst)[0] = z; ((uint4*)dst)[1] = z;
      }
    }
    wait_async0();
    __syncthreads();

    AFrag a[2], b[4];
    const int abase = 8 * (lane >> 4);
    const int arow0 = waveM + (lane & 15);
#pragma unroll
    for (int tm = 0; tm < 2; ++tm) {
      const u16* row = As + (arow0 + tm * 16) * 32;
      a[tm].q[0] = *(const uint4*)(row + abase);
      a[tm].q[1] = *(const uint4*)(row + abase + 16);
    }
    const int bbase = 8 * (lane >> 4);
#pragma unroll
    for (int tn = 0; tn < 4; ++tn) {
      const u16* row = Bs + (waveN + tn * 16 + (lane & 15)) * 32;
      b[tn].q[0] = *(const uint4*)(row + bbase);
      b[tn].q[1] = *(const uint4*)(row + bbase + 16);
    }
#pragma unroll
    for (int tm = 0; tm < 2; ++tm)
#pragma unroll
      for (int tn = 0; tn < 4; ++tn)
        acc[tm][tn].v = wmma_bf16(a[tm].v, b[tn].v, acc[tm][tn].v);
    __syncthreads();
  }

  const int mloc = (lane >> 4) * 8, nloc = lane & 15;
#pragma unroll
  for (int tm = 0; tm < 2; ++tm)
#pragma unroll
    for (int tn = 0; tn < 4; ++tn) {
      int n = blockN + waveN + tn * 16 + nloc;
      if (n < NSEMEME) {
#pragma unroll
        for (int i = 0; i < 8; ++i) {
          int m = blockM + waveM + tm * 16 + mloc + i;
          float v = acc[tm][tn].f[i] + gbias[n];
          gate[(size_t)m * NSEMEME + n] = 1.0f / (1.0f + expf(-v));
        }
      }
    }
}

// ---------------------------------------------------------------- transed[k] = x @ multi_trans[k] -> bf16
__global__ __launch_bounds__(256)
void transed_kernel(const u16* __restrict__ xbf, const float* __restrict__ mt,
                    u16* __restrict__ tr) {
  __shared__ u16 As[128 * 32];
  __shared__ u16 Bs[128 * 32];
  const int tid = threadIdx.x, lane = tid & 31, wid = tid >> 5;
  const int waveM = (wid & 3) * 32, waveN = (wid >> 2) * 64;
  const int blockM = blockIdx.y * 128, blockN = blockIdx.x * 128;
  const int kb = blockIdx.z;
  const int r = tid >> 1, c0 = (tid & 1) * 16;
  const unsigned ldsa = (unsigned)(uintptr_t)(As + r * 32 + c0);
  const u16* arow = xbf + (size_t)(blockM + r) * NHID + c0;
  // transposed-B staging: each thread converts pairs along h (contiguous in LDS [d][h])
  const int h0 = (tid >> 4) * 2;        // 0,2,..,30
  const int d0 = (tid & 15) * 8;        // 0,8,..,120

  CFrag acc[2][4];
#pragma unroll
  for (int tm = 0; tm < 2; ++tm)
#pragma unroll
    for (int tn = 0; tn < 4; ++tn)
#pragma unroll
      for (int i = 0; i < 8; ++i) acc[tm][tn].f[i] = 0.0f;

  for (int k0 = 0; k0 < NHID; k0 += 32) {
    // stage A: async bf16 copy
    async_copy32(ldsa, arow + k0);
    // stage B transposed: LDS[d][h], source mt[kb][h][d]; pack (h0, h0+1) pairs
    {
      const float* rowA = mt + (size_t)kb * NHID * NHID + (size_t)(k0 + h0) * NHID + blockN + d0;
      const float* rowB = rowA + NHID;
      float4 a0 = ((const float4*)rowA)[0], a1 = ((const float4*)rowA)[1];
      float4 b0 = ((const float4*)rowB)[0], b1 = ((const float4*)rowB)[1];
      float av[8] = {a0.x, a0.y, a0.z, a0.w, a1.x, a1.y, a1.z, a1.w};
      float bv[8] = {b0.x, b0.y, b0.z, b0.w, b1.x, b1.y, b1.z, b1.w};
#pragma unroll
      for (int j = 0; j < 8; ++j)
        *(unsigned*)(Bs + (d0 + j) * 32 + h0) = pk2(av[j], bv[j]);
    }
    wait_async0();
    __syncthreads();

    AFrag a[2], b[4];
    const int abase = 8 * (lane >> 4);
    const int arow0 = waveM + (lane & 15);
#pragma unroll
    for (int tm = 0; tm < 2; ++tm) {
      const u16* row = As + (arow0 + tm * 16) * 32;
      a[tm].q[0] = *(const uint4*)(row + abase);
      a[tm].q[1] = *(const uint4*)(row + abase + 16);
    }
    const int bbase = 8 * (lane >> 4);
#pragma unroll
    for (int tn = 0; tn < 4; ++tn) {
      const u16* row = Bs + (waveN + tn * 16 + (lane & 15)) * 32;
      b[tn].q[0] = *(const uint4*)(row + bbase);
      b[tn].q[1] = *(const uint4*)(row + bbase + 16);
    }
#pragma unroll
    for (int tm = 0; tm < 2; ++tm)
#pragma unroll
      for (int tn = 0; tn < 4; ++tn)
        acc[tm][tn].v = wmma_bf16(a[tm].v, b[tn].v, acc[tm][tn].v);
    __syncthreads();
  }

  const int mloc = (lane >> 4) * 8, nloc = lane & 15;
#pragma unroll
  for (int tm = 0; tm < 2; ++tm)
#pragma unroll
    for (int tn = 0; tn < 4; ++tn) {
      int n = blockN + waveN + tn * 16 + nloc;
#pragma unroll
      for (int i = 0; i < 8; ++i) {
        int m = blockM + waveM + tm * 16 + mloc + i;
        tr[((size_t)kb * NB + m) * NHID + n] = f2bf(acc[tm][tn].f[i]);
      }
    }
}

// ---------------------------------------------------------------- fused big GEMM:
// logits[b,s] = sum_k (transed[k,b,:] . enc[widx[s],:]) * coeff(k,b,s) + sense_bias[s]
__global__ __launch_bounds__(256)
void big_kernel(const u16* __restrict__ trbf, const float* __restrict__ enc,
                const int* __restrict__ widx, const float* __restrict__ gate,
                const float* __restrict__ wsm, const int* __restrict__ sidx,
                const float* __restrict__ sbias, float* __restrict__ logits) {
  __shared__ u16 As[128 * 32];
  __shared__ u16 Bs[128 * 32];
  const int tid = threadIdx.x, lane = tid & 31, wid = tid >> 5;
  const int waveM = (wid & 3) * 32, waveN = (wid >> 2) * 64;
  const int blockM = blockIdx.y * 128, blockN = blockIdx.x * 128;
  const int r = tid >> 1, c0 = (tid & 1) * 16;
  const int4* sidx4 = (const int4*)sidx;

  // gather index for the B row this thread stages (loop-invariant)
  const int sg = blockN + r;
  const int wrow = (sg < NSENSE) ? widx[sg] : -1;

  const unsigned ldsa = (unsigned)(uintptr_t)(As + r * 32 + c0);

  CFrag fin[2][4];
#pragma unroll
  for (int tm = 0; tm < 2; ++tm)
#pragma unroll
    for (int tn = 0; tn < 4; ++tn)
#pragma unroll
      for (int i = 0; i < 8; ++i) fin[tm][tn].f[i] = 0.0f;

  const int mloc = (lane >> 4) * 8, nloc = lane & 15;

  for (int kb = 0; kb < NBASIS; ++kb) {
    CFrag work[2][4];
#pragma unroll
    for (int tm = 0; tm < 2; ++tm)
#pragma unroll
      for (int tn = 0; tn < 4; ++tn)
#pragma unroll
        for (int i = 0; i < 8; ++i) work[tm][tn].f[i] = 0.0f;

    const u16* arow = trbf + ((size_t)kb * NB + blockM + r) * NHID + c0;

    for (int k0 = 0; k0 < NHID; k0 += 32) {
      // stage A: async bf16 copy of transed rows (32B/thread)
      async_copy32(ldsa, arow + k0);
      { // stage B: gather encoder rows, f32 -> bf16 (packed)
        u16* dst = Bs + r * 32 + c0;
        if (wrow >= 0) {
          cvt16_store((const float4*)(enc + (size_t)wrow * NHID + k0 + c0), dst);
        } else {
          uint4 z = {0, 0, 0, 0};
          ((uint4*)dst)[0] = z; ((uint4*)dst)[1] = z;
        }
      }
      wait_async0();
      __syncthreads();

      AFrag a[2], b[4];
      const int abase = 8 * (lane >> 4);
      const int arow0 = waveM + (lane & 15);
#pragma unroll
      for (int tm = 0; tm < 2; ++tm) {
        const u16* row = As + (arow0 + tm * 16) * 32;
        a[tm].q[0] = *(const uint4*)(row + abase);
        a[tm].q[1] = *(const uint4*)(row + abase + 16);
      }
      const int bbase = 8 * (lane >> 4);
#pragma unroll
      for (int tn = 0; tn < 4; ++tn) {
        const u16* row = Bs + (waveN + tn * 16 + (lane & 15)) * 32;
        b[tn].q[0] = *(const uint4*)(row + bbase);
        b[tn].q[1] = *(const uint4*)(row + bbase + 16);
      }
#pragma unroll
      for (int tm = 0; tm < 2; ++tm)
#pragma unroll
        for (int tn = 0; tn < 4; ++tn)
          work[tm][tn].v = wmma_bf16(a[tm].v, b[tn].v, work[tm][tn].v);
      __syncthreads();
    }

    // epilogue for this basis: fin += work * coeff(kb, b, s)
#pragma unroll
    for (int tm = 0; tm < 2; ++tm)
#pragma unroll
      for (int tn = 0; tn < 4; ++tn) {
        int s = blockN + waveN + tn * 16 + nloc;
        if (s < NSENSE) {
          int4 id4 = sidx4[s];
          float w0 = wsm[id4.x * 3 + kb], w1 = wsm[id4.y * 3 + kb];
          float w2 = wsm[id4.z * 3 + kb], w3 = wsm[id4.w * 3 + kb];
#pragma unroll
          for (int i = 0; i < 8; ++i) {
            int bb = blockM + waveM + tm * 16 + mloc + i;
            const float* grow = gate + (size_t)bb * NSEMEME;
            float cf = grow[id4.x] * w0 + grow[id4.y] * w1 +
                       grow[id4.z] * w2 + grow[id4.w] * w3;
            fin[tm][tn].f[i] += work[tm][tn].f[i] * cf;
          }
        }
      }
  }

  // final store: logits = fin + sense_bias
#pragma unroll
  for (int tm = 0; tm < 2; ++tm)
#pragma unroll
    for (int tn = 0; tn < 4; ++tn) {
      int s = blockN + waveN + tn * 16 + nloc;
      if (s < NSENSE) {
        float sb = sbias[s];
#pragma unroll
        for (int i = 0; i < 8; ++i) {
          int bb = blockM + waveM + tm * 16 + mloc + i;
          logits[(size_t)bb * NSENSE + s] = fin[tm][tn].f[i] + sb;
        }
      }
    }
}

// ---------------------------------------------------------------- per-row softmax stats
__global__ __launch_bounds__(256)
void row_stats_kernel(const float* __restrict__ logits, float* __restrict__ stats) {
  __shared__ float red[256];
  const int b = blockIdx.x, tid = threadIdx.x;
  const float* row = logits + (size_t)b * NSENSE;
  float mx = -3.4e38f;
  for (int s = tid; s < NSENSE; s += 256) mx = fmaxf(mx, row[s]);
  red[tid] = mx; __syncthreads();
  for (int w = 128; w > 0; w >>= 1) {
    if (tid < w) red[tid] = fmaxf(red[tid], red[tid + w]);
    __syncthreads();
  }
  float rmax = red[0]; __syncthreads();
  float sum = 0.0f;
  for (int s = tid; s < NSENSE; s += 256) sum += expf(row[s] - rmax);
  red[tid] = sum; __syncthreads();
  for (int w = 128; w > 0; w >>= 1) {
    if (tid < w) red[tid] += red[tid + w];
    __syncthreads();
  }
  if (tid == 0) { stats[2 * b] = rmax; stats[2 * b + 1] = red[0]; }
}

// ---------------------------------------------------------------- zero d_out (vectorized)
__global__ __launch_bounds__(256)
void zero_kernel(float4* __restrict__ out) {
  out[blockIdx.x * 256 + threadIdx.x] = float4{0.0f, 0.0f, 0.0f, 0.0f};
}

// ---------------------------------------------------------------- scatter softmax probs by word idx
__global__ __launch_bounds__(256)
void scatter_kernel(const float* __restrict__ logits, const float* __restrict__ stats,
                    const int* __restrict__ widx, float* __restrict__ out) {
  int i = blockIdx.x * 256 + threadIdx.x;   // 0 .. 256*50000-1
  int b = i / NSENSE;
  int s = i - b * NSENSE;
  float p = expf(logits[i] - stats[2 * b]) / stats[2 * b + 1];
  atomicAdd(out + (size_t)b * NTOKEN + widx[s], p);
}

// ---------------------------------------------------------------- out = log(clip(out, 1e-8))
__global__ __launch_bounds__(256)
void log_kernel(float4* __restrict__ out) {
  int i = blockIdx.x * 256 + threadIdx.x;
  float4 v = out[i];
  v.x = logf(fmaxf(v.x, 1e-8f));
  v.y = logf(fmaxf(v.y, 1e-8f));
  v.z = logf(fmaxf(v.z, 1e-8f));
  v.w = logf(fmaxf(v.w, 1e-8f));
  out[i] = v;
}

// ================================================================ host launcher
extern "C" void kernel_launch(void* const* d_in, const int* in_sizes, int n_in,
                              void* d_out, int out_size, void* d_ws, size_t ws_size,
                              hipStream_t stream) {
  const float* x     = (const float*)d_in[0];   // (4,64,1024) -> (256,1024)
  const float* enc   = (const float*)d_in[1];   // (50000,1024)
  const float* wg    = (const float*)d_in[2];   // (2000,1024)
  const float* gbias = (const float*)d_in[3];   // (2000,)
  const float* mt    = (const float*)d_in[4];   // (3,1024,1024)
  const float* mw    = (const float*)d_in[5];   // (3,1,2000)
  const float* sbias = (const float*)d_in[6];   // (1,50000)
  const int*   widx  = (const int*)d_in[7];     // (50000,)
  const int*   sidx  = (const int*)d_in[8];     // (50000,4)
  float*       out   = (float*)d_out;           // (256,50000)

  char* W = (char*)d_ws;
  size_t off = 0;
  auto take = [&](size_t bytes) -> void* {
    void* p = W + off;
    off = (off + bytes + 255) & ~(size_t)255;
    return p;
  };
  float* wsm    = (float*)take((size_t)NSEMEME * 3 * sizeof(float));           // 24 KB
  float* gate   = (float*)take((size_t)NB * NSEMEME * sizeof(float));          // 2.05 MB
  u16*   trbf   = (u16*)take((size_t)NBASIS * NB * NHID * sizeof(u16));        // 1.57 MB
  u16*   xbf    = (u16*)take((size_t)NB * NHID * sizeof(u16));                 // 512 KB
  float* logits = (float*)take((size_t)NB * NSENSE * sizeof(float));           // 51.2 MB
  float* stats  = (float*)take((size_t)NB * 2 * sizeof(float));                // 2 KB
  (void)ws_size; (void)in_sizes; (void)n_in; (void)out_size;

  xbf_kernel<<<(NB * NHID) / (256 * 16), 256, 0, stream>>>(x, xbf);
  wsm_kernel<<<(NSEMEME + 255) / 256, 256, 0, stream>>>(mw, wsm);
  gate_kernel<<<dim3((NSEMEME + 127) / 128, NB / 128), 256, 0, stream>>>(xbf, wg, gbias, gate);
  transed_kernel<<<dim3(NHID / 128, NB / 128, NBASIS), 256, 0, stream>>>(xbf, mt, trbf);
  big_kernel<<<dim3((NSENSE + 127) / 128, NB / 128), 256, 0, stream>>>(
      trbf, enc, widx, gate, wsm, sidx, sbias, logits);
  row_stats_kernel<<<NB, 256, 0, stream>>>(logits, stats);
  zero_kernel<<<(NB * NTOKEN) / (256 * 4), 256, 0, stream>>>((float4*)out);
  scatter_kernel<<<(NB * NSENSE) / 256, 256, 0, stream>>>(logits, stats, widx, out);
  log_kernel<<<(NB * NTOKEN) / (256 * 4), 256, 0, stream>>>((float4*)out);
}
